// DS_Fusion_56495999811926
// MI455X (gfx1250) — matrix operand order repaired and checked by reading, not compiled, not run on hardware
//
#include <hip/hip_runtime.h>
#include <hip/hip_bf16.h>
#include <math.h>

typedef _Float16 f16x8  __attribute__((ext_vector_type(8)));
typedef _Float16 f16x16 __attribute__((ext_vector_type(16)));
typedef float    f32x8  __attribute__((ext_vector_type(8)));

#define HW      16384
#define TILE    64
#define NTHR    128
#define XS      72      // row stride (halves) for X0/X1/LA/KK state buffers
#define QS      96      // row stride for Q / V buffers
#define NFRAG   46

// fragment directory (units of 512 halves)
#define FR_RB1  0       // 4 frags : 48->24
#define FR_RB2  4       // 3 frags : 24->48
#define FR_Q    7       // 6 frags : 48->48
#define FR_V    13      // 6 frags : 48->48
#define FR_K1   19      // 4 frags : 48->24
#define FR_K2   23      // 4 frags : 48->24
#define FR_CF   27      // 12 frags: 96->48 (split: k-steps 0,1 read X0; 2,3 read X1)
#define FR_E1   39      // 4 frags : 48->24
#define FR_E2   43      // 3 frags : 24->48

// ---- WMMA fragment loads (layouts per CDNA5 ISA 7.12.2, wave32) ----
__device__ __forceinline__ f16x16 load_a_frag(const _Float16* buf, int stride,
                                              int pixBase, int kBase, int lane) {
  // A 16x32 f16: lane<16 -> M=lane, K: e0..7 = 0..7, e8..15 = 16..23
  //              lane>=16 -> M=lane-16, K: e0..7 = 8..15, e8..15 = 24..31
  const _Float16* p = buf + (pixBase + (lane & 15)) * stride + kBase + ((lane >> 4) << 3);
  f16x8 lo = *(const f16x8*)p;
  f16x8 hi = *(const f16x8*)(p + 16);
  return __builtin_shufflevector(lo, hi, 0,1,2,3,4,5,6,7,8,9,10,11,12,13,14,15);
}

__device__ __forceinline__ f16x16 load_b_frag(const _Float16* frag, int lane) {
  // B 32x16 f16 pre-swizzled: [lane][e] contiguous, N=lane&15, K=(e&7)+8*(lane>>4)+16*(e>>3)
  const _Float16* p = frag + lane * 16;
  f16x8 lo = *(const f16x8*)p;
  f16x8 hi = *(const f16x8*)(p + 8);
  return __builtin_shufflevector(lo, hi, 0,1,2,3,4,5,6,7,8,9,10,11,12,13,14,15);
}

// MODE: 0=none 1=relu 2=gelu(exact erf) 3=residual+relu
template<int KSTEPS, int NCHUNKS, int KHALF, int MODE>
__device__ __forceinline__ void conv_tile(const _Float16* in0, const _Float16* in1, int inStride,
                                          _Float16* outBuf, int outStride,
                                          const _Float16* frag, const float* sb,
                                          const _Float16* resBuf, int resStride,
                                          int pixBase, int lane) {
#pragma unroll
  for (int nb = 0; nb < NCHUNKS; ++nb) {
    f32x8 acc = {0.f,0.f,0.f,0.f,0.f,0.f,0.f,0.f};
#pragma unroll
    for (int kb = 0; kb < KSTEPS; ++kb) {
      const _Float16* ib = (kb < KHALF) ? in0 : in1;
      int kBase = ((kb < KHALF) ? kb : (kb - KHALF)) * 32;
      f16x16 a = load_a_frag(ib, inStride, pixBase, kBase, lane);
      f16x16 b = load_b_frag(frag + (kb * NCHUNKS + nb) * 512, lane);
      acc = __builtin_amdgcn_wmma_f32_16x16x32_f16(false, a, false, b,
                                                   (short)0, acc, false, false);
    }
    int n = nb * 16 + (lane & 15);
    float s = sb[n];
    float t = sb[64 + n];
    int rowTop = pixBase + ((lane >> 4) << 3);
#pragma unroll
    for (int j = 0; j < 8; ++j) {
      int p = rowTop + j;                 // D: M = j + 8*(lane>>4), N = lane&15
      float v = acc[j];
      if constexpr (MODE == 3) v += (float)resBuf[p * resStride + n];
      v = v * s + t;
      if constexpr (MODE == 1 || MODE == 3) v = fmaxf(v, 0.f);
      if constexpr (MODE == 2) v = 0.5f * v * (1.f + erff(v * 0.70710678118654752f));
      outBuf[p * outStride + n] = (_Float16)v;
    }
  }
}

__global__ __launch_bounds__(NTHR) void fused_dsattn_kernel(
    const float* __restrict__ x0, const float* __restrict__ x1,
    const float* __restrict__ rb_w1, const float* __restrict__ rb_b1,
    const float* __restrict__ rb_bn_g, const float* __restrict__ rb_bn_b,
    const float* __restrict__ rb_bn_m, const float* __restrict__ rb_bn_v,
    const float* __restrict__ rb_w2, const float* __restrict__ rb_b2,
    const float* __restrict__ bn_g, const float* __restrict__ bn_b,
    const float* __restrict__ bn_m, const float* __restrict__ bn_v,
    const float* __restrict__ q_w, const float* __restrict__ q_b,
    const float* __restrict__ q_bn_g, const float* __restrict__ q_bn_b,
    const float* __restrict__ q_bn_m, const float* __restrict__ q_bn_v,
    const float* __restrict__ v_w, const float* __restrict__ v_b,
    const float* __restrict__ v_bn_g, const float* __restrict__ v_bn_b,
    const float* __restrict__ v_bn_m, const float* __restrict__ v_bn_v,
    const float* __restrict__ k1_w, const float* __restrict__ k1_b,
    const float* __restrict__ k1_bn_g, const float* __restrict__ k1_bn_b,
    const float* __restrict__ k1_bn_m, const float* __restrict__ k1_bn_v,
    const float* __restrict__ k2_w, const float* __restrict__ k2_b,
    const float* __restrict__ k2_bn_g, const float* __restrict__ k2_bn_b,
    const float* __restrict__ k2_bn_m, const float* __restrict__ k2_bn_v,
    const float* __restrict__ cf_w, const float* __restrict__ cf_b,
    const float* __restrict__ cf_bn_g, const float* __restrict__ cf_bn_b,
    const float* __restrict__ cf_bn_m, const float* __restrict__ cf_bn_v,
    const float* __restrict__ emb_w1, const float* __restrict__ emb_b1,
    const float* __restrict__ emb_w2, const float* __restrict__ emb_b2,
    float* __restrict__ out) {

  __shared__ __align__(16) _Float16 sX0[TILE * XS];
  __shared__ __align__(16) _Float16 sX1[TILE * XS];
  __shared__ __align__(16) _Float16 sLA[TILE * XS];
  __shared__ __align__(16) _Float16 sKK[TILE * XS];     // cf-out / kk / refs / emb-in
  __shared__ __align__(16) _Float16 sQ[TILE * QS];
  __shared__ __align__(16) _Float16 sV[TILE * QS];      // r-temp / values / emb-hidden
  __shared__ __align__(16) _Float16 sFrag[NFRAG * 512];
  __shared__ float sSB[9 * 128];                        // per conv: [0..63]=scale [64..127]=bias

  const int tid = threadIdx.x;
  const int lane = tid & 31;
  const int pixBase = (tid >> 5) << 4;                  // wave-owned 16-pixel subtile
  const int bIdx = blockIdx.x >> 8;                     // 256 tiles per image
  const int hwBase = (blockIdx.x & 255) << 6;
  const size_t base0 = (size_t)bIdx * 48 * HW + hwBase;

  // ---- fold BN+bias into per-channel scale/bias (pad channels -> 0) ----
  for (int idx = tid; idx < 9 * 64; idx += NTHR) {
    int conv = idx >> 6, n = idx & 63;
    float s = 0.f, t = 0.f;
    const float *g=nullptr,*bb=nullptr,*mm=nullptr,*vv=nullptr,*cb=nullptr; int cout=0;
    switch (conv) {
      case 0: g=rb_bn_g; bb=rb_bn_b; mm=rb_bn_m; vv=rb_bn_v; cb=rb_b1; cout=24; break;
      case 1: g=bn_g;    bb=bn_b;    mm=bn_m;    vv=bn_v;    cb=rb_b2; cout=48; break;
      case 2: g=q_bn_g;  bb=q_bn_b;  mm=q_bn_m;  vv=q_bn_v;  cb=q_b;   cout=48; break;
      case 3: g=v_bn_g;  bb=v_bn_b;  mm=v_bn_m;  vv=v_bn_v;  cb=v_b;   cout=48; break;
      case 4: g=k1_bn_g; bb=k1_bn_b; mm=k1_bn_m; vv=k1_bn_v; cb=k1_b;  cout=24; break;
      case 5: g=k2_bn_g; bb=k2_bn_b; mm=k2_bn_m; vv=k2_bn_v; cb=k2_b;  cout=24; break;
      case 6: g=cf_bn_g; bb=cf_bn_b; mm=cf_bn_m; vv=cf_bn_v; cb=cf_b;  cout=48; break;
      case 7: if (n < 24) { s = 1.f; t = emb_b1[n]; } break;
      case 8: if (n < 48) { s = 1.f; t = emb_b2[n]; } break;
    }
    if (g && n < cout) {
      float sc = g[n] / sqrtf(vv[n] + 1e-5f);
      s = sc; t = cb[n] * sc + bb[n] - mm[n] * sc;
    }
    sSB[conv * 128 + n] = s;
    sSB[conv * 128 + 64 + n] = t;
  }

  // ---- pre-swizzle all weights into B-fragment layout (f16) ----
  for (int idx = tid; idx < NFRAG * 512; idx += NTHR) {
    int f = idx >> 9, r = idx & 511, ln = r >> 4, e = r & 15;
    int Kl = (e & 7) + ((ln >> 4) << 3) + ((e >> 3) << 4);  // K within 32-step
    int N  = ln & 15;
    const float* w = nullptr; int Cin = 0, Cout = 0, kb = 0, nb = 0; bool cf = false;
    if      (f < 4)  { w=rb_w1;  Cin=48; Cout=24; kb=f>>1;       nb=f&1; }
    else if (f < 7)  { w=rb_w2;  Cin=24; Cout=48; kb=0;          nb=f-4; }
    else if (f < 13) { int g=f-7;  w=q_w;    Cin=48; Cout=48; kb=g/3;  nb=g%3; }
    else if (f < 19) { int g=f-13; w=v_w;    Cin=48; Cout=48; kb=g/3;  nb=g%3; }
    else if (f < 23) { int g=f-19; w=k1_w;   Cin=48; Cout=24; kb=g>>1; nb=g&1; }
    else if (f < 27) { int g=f-23; w=k2_w;   Cin=48; Cout=24; kb=g>>1; nb=g&1; }
    else if (f < 39) { int g=f-27; w=cf_w;   Cin=96; Cout=48; kb=g/3;  nb=g%3; cf=true; }
    else if (f < 43) { int g=f-39; w=emb_w1; Cin=48; Cout=24; kb=g>>1; nb=g&1; }
    else             { int g=f-43; w=emb_w2; Cin=24; Cout=48; kb=0;    nb=g;   }
    int n = nb * 16 + N;
    float val = 0.f;
    if (cf) {                              // k-steps 0,1: x0 half; 2,3: x1 half
      int part = kb >> 1;
      int i = (kb & 1) * 32 + Kl;
      if (i < 48 && n < Cout) val = w[n * 96 + part * 48 + i];
    } else {
      int i = kb * 32 + Kl;
      if (i < Cin && n < Cout) val = w[n * Cin + i];
    }
    sFrag[(f << 9) + (ln << 4) + e] = (_Float16)val;
  }

  // ---- load x0/x1 tiles -> f16 state (K-pad channels zeroed) ----
  for (int idx = tid; idx < TILE * XS; idx += NTHR) {
    int p = idx & 63, c = idx >> 6;        // coalesced over p per channel
    float a0 = (c < 48) ? x0[base0 + (size_t)c * HW + p] : 0.f;
    float a1 = (c < 48) ? x1[base0 + (size_t)c * HW + p] : 0.f;
    sX0[p * XS + c] = (_Float16)a0;
    sX1[p * XS + c] = (_Float16)a1;
  }
  for (int idx = tid; idx < TILE * 24; idx += NTHR) {    // zero LA/KK pad channels
    int p = idx / 24, c = 48 + idx % 24;
    sLA[p * XS + c] = (_Float16)0.f;
    sKK[p * XS + c] = (_Float16)0.f;
  }
  __syncthreads();

  for (int k = 0; k < 4; ++k) {
    // residual refinement (wave-local WMMA chain)
    conv_tile<2,2,4,1>(sX0, sX0, XS, sV, QS, sFrag + FR_RB1*512, sSB + 0*128, nullptr, 0, pixBase, lane);
    conv_tile<1,3,4,3>(sV,  sV,  QS, sX0, XS, sFrag + FR_RB2*512, sSB + 1*128, sX0, XS, pixBase, lane);
    conv_tile<2,2,4,1>(sX1, sX1, XS, sV, QS, sFrag + FR_RB1*512, sSB + 0*128, nullptr, 0, pixBase, lane);
    conv_tile<1,3,4,3>(sV,  sV,  QS, sX1, XS, sFrag + FR_RB2*512, sSB + 1*128, sX1, XS, pixBase, lane);

    if (k == 0) {  // cf over concat(x0,x1) as 4 K-steps, then emb -> la_res
      conv_tile<4,3,2,1>(sX0, sX1, XS, sKK, XS, sFrag + FR_CF*512, sSB + 6*128, nullptr, 0, pixBase, lane);
      conv_tile<2,2,4,2>(sKK, sKK, XS, sV,  QS, sFrag + FR_E1*512, sSB + 7*128, nullptr, 0, pixBase, lane);
      conv_tile<1,3,4,0>(sV,  sV,  QS, sLA, XS, sFrag + FR_E2*512, sSB + 8*128, nullptr, 0, pixBase, lane);
    }

    // query / value / keys
    conv_tile<2,3,4,0>(sX0, sX0, XS, sQ,      QS, sFrag + FR_Q*512,  sSB + 2*128, nullptr, 0, pixBase, lane);
    conv_tile<2,3,4,0>(sX1, sX1, XS, sQ + 48, QS, sFrag + FR_Q*512,  sSB + 2*128, nullptr, 0, pixBase, lane);
    conv_tile<2,3,4,0>(sX0, sX0, XS, sV,      QS, sFrag + FR_V*512,  sSB + 3*128, nullptr, 0, pixBase, lane);
    conv_tile<2,3,4,0>(sX1, sX1, XS, sV + 48, QS, sFrag + FR_V*512,  sSB + 3*128, nullptr, 0, pixBase, lane);
    conv_tile<2,2,4,0>(sLA, sLA, XS, sKK,      XS, sFrag + FR_K1*512, sSB + 4*128, nullptr, 0, pixBase, lane);
    conv_tile<2,2,4,0>(sLA, sLA, XS, sKK + 24, XS, sFrag + FR_K2*512, sSB + 5*128, nullptr, 0, pixBase, lane);

    __syncthreads();
    // per-pixel 4-way softmax attention; refs written over kk in-place
    {
      int p = tid & 63, which = tid >> 6;               // which: 0 -> k1 ref, 1 -> k2 ref
      const _Float16* kk = sKK + p * XS + which * 24;
      const _Float16* qq = sQ + p * QS;
      const _Float16* vv = sV + p * QS;
      float l[4];
#pragma unroll
      for (int m = 0; m < 4; ++m) {
        float a = 0.f;
#pragma unroll
        for (int c = 0; c < 24; ++c) a += (float)kk[c] * (float)qq[m * 24 + c];
        l[m] = a;
      }
      float mx = fmaxf(fmaxf(l[0], l[1]), fmaxf(l[2], l[3]));
      float e0 = expf(l[0]-mx), e1 = expf(l[1]-mx), e2 = expf(l[2]-mx), e3 = expf(l[3]-mx);
      float inv = 1.f / (e0 + e1 + e2 + e3);
      e0 *= inv; e1 *= inv; e2 *= inv; e3 *= inv;
      _Float16* rf = sKK + p * XS + which * 24;
#pragma unroll
      for (int c = 0; c < 24; ++c) {
        float r = e0 * (float)vv[c] + e1 * (float)vv[24 + c]
                + e2 * (float)vv[48 + c] + e3 * (float)vv[72 + c];
        rf[c] = (_Float16)r;
      }
      if (which == 0)
        for (int j = 0; j < 16; ++j) sKK[p * XS + 48 + j] = (_Float16)0.f;
    }
    __syncthreads();

    // la_res = emb(concat(refs))
    conv_tile<2,2,4,2>(sKK, sKK, XS, sV,  QS, sFrag + FR_E1*512, sSB + 7*128, nullptr, 0, pixBase, lane);
    conv_tile<1,3,4,0>(sV,  sV,  QS, sLA, XS, sFrag + FR_E2*512, sSB + 8*128, nullptr, 0, pixBase, lane);
    __syncthreads();
  }

  // write out [B,48,H,W] in f32
  for (int idx = tid; idx < TILE * 48; idx += NTHR) {
    int p = idx & 63, c = idx >> 6;
    out[base0 + (size_t)c * HW + p] = (float)sLA[p * XS + c];
  }
}

extern "C" void kernel_launch(void* const* d_in, const int* in_sizes, int n_in,
                              void* d_out, int out_size, void* d_ws, size_t ws_size,
                              hipStream_t stream) {
  (void)in_sizes; (void)n_in; (void)out_size; (void)d_ws; (void)ws_size;
  const float* a[48];
  for (int i = 0; i < 48; ++i) a[i] = (const float*)d_in[i];
  dim3 grid(16 * (HW / TILE));   // 16 images * 256 tiles = 4096 blocks
  dim3 block(NTHR);
  fused_dsattn_kernel<<<grid, block, 0, stream>>>(
      a[0], a[1], a[2], a[3], a[4], a[5], a[6], a[7], a[8], a[9],
      a[10], a[11], a[12], a[13], a[14], a[15], a[16], a[17], a[18], a[19],
      a[20], a[21], a[22], a[23], a[24], a[25], a[26], a[27], a[28], a[29],
      a[30], a[31], a[32], a[33], a[34], a[35], a[36], a[37], a[38], a[39],
      a[40], a[41], a[42], a[43], a[44], a[45], a[46], a[47],
      (float*)d_out);
}
